// SelfAttention_84645215469985
// MI455X (gfx1250) — matrix operand order repaired and checked by reading, not compiled, not run on hardware
//
#include <hip/hip_runtime.h>
#include <math.h>

typedef __bf16 bf16_t;
typedef __attribute__((ext_vector_type(16))) __bf16 v16bf;
typedef __attribute__((ext_vector_type(8)))  __bf16 v8bf;
typedef __attribute__((ext_vector_type(8)))  float  v8f;

#define B_  2
#define H_  16
#define S_  2048
#define D_  128
#define BR  128   // query rows per workgroup
#define BC  32    // key rows per tile
#define NT  (S_ / BC)
#define NW  8     // waves per workgroup (wave32)
#define QSTR 136  // LDS row stride (bf16) for Q  (conflict-free b128 frag loads)
#define KSTR 40   // LDS row stride for K^T
#define VSTR 136  // LDS row stride for V
#define PSTR 40   // LDS row stride for per-wave P tile

__device__ __forceinline__ v16bf pack16(v8bf lo, v8bf hi) {
    v16bf r;
#pragma unroll
    for (int i = 0; i < 8; ++i) { r[i] = lo[i]; r[i + 8] = hi[i]; }
    return r;
}

// ---------------------------------------------------------------------------
// mean-V helper kernels: ws[b*D+d] = sum_{h,s} v[b,h,s,d]
// ---------------------------------------------------------------------------
__global__ void zero_ws_kernel(float* ws) { ws[threadIdx.x] = 0.0f; }

__global__ __launch_bounds__(256) void sumv_kernel(const float* __restrict__ v,
                                                   float* __restrict__ ws) {
    // grid: (S/256, H, B)
    const int d    = threadIdx.x & (D_ - 1);
    const int half = threadIdx.x >> 7;           // 0 or 1
    const int b = blockIdx.z, h = blockIdx.y;
    const size_t base = (((size_t)b * H_ + h) * S_ + (size_t)blockIdx.x * 256) * D_;
    float acc = 0.0f;
    for (int s = half; s < 256; s += 2) acc += v[base + (size_t)s * D_ + d];
    atomicAdd(&ws[b * D_ + d], acc);
}

__global__ __launch_bounds__(256) void initout_kernel(const float* __restrict__ ws,
                                                      float* __restrict__ out) {
    // grid: B*S*D/256
    const size_t i = (size_t)blockIdx.x * 256 + threadIdx.x;
    const int d = (int)(i % D_);
    const int b = (int)(i / ((size_t)S_ * D_));
    out[i] = ws[b * D_ + d] * ((float)(H_ - 1) / (float)S_);
}

// ---------------------------------------------------------------------------
// Flash attention over one (b, h, 128-row query block); heads summed into out
// via global f32 atomics. K/V tiles are double-buffered in LDS: next tile's
// global loads are issued before the WMMA phase so VMEM latency hides under
// matrix math; LOADcnt is only waited at the cvt+ds_store stage afterwards.
// ---------------------------------------------------------------------------
__global__ __launch_bounds__(256) void fa_bf16_wmma_kernel(
    const float* __restrict__ q, const float* __restrict__ k,
    const float* __restrict__ v, float* __restrict__ out) {

    __shared__ alignas(16) bf16_t Qs[BR * QSTR];          // ~34.8 KB
    __shared__ alignas(16) bf16_t Kts[2][D_ * KSTR];      // ~20.5 KB (transposed [d][r])
    __shared__ alignas(16) bf16_t Vs[2][BC * VSTR];       // ~17.4 KB (row-major [r][d])
    __shared__ alignas(16) bf16_t Ps[NW][16 * PSTR];      // ~10.2 KB per-wave P tiles

    const int t    = threadIdx.x;
    const int w    = t >> 5;
    const int lane = t & 31;
    const int qblk = blockIdx.x;
    const int h    = blockIdx.y;
    const int b    = blockIdx.z;

    const float scale = 0.08838834764831845f;   // 1/sqrt(128)

    const float* qbase = q + (((size_t)b * H_ + h) * S_ + (size_t)qblk * BR) * D_;
    const float* kbase = k + (((size_t)b * H_ + h) * S_) * D_;
    const float* vbase = v + (((size_t)b * H_ + h) * S_) * D_;

    // ---- stage Q block to LDS as bf16, folding in softmax scale ----
#pragma unroll
    for (int i = 0; i < (BR * D_) / (256 * 4); ++i) {   // 16 iters of float4
        const int lin = (t + i * 256) * 4;
        const int r = lin / D_, c = lin % D_;
        const float4 f = *(const float4*)(qbase + (size_t)r * D_ + c);
        bf16_t* d = &Qs[r * QSTR + c];
        d[0] = (bf16_t)(f.x * scale); d[1] = (bf16_t)(f.y * scale);
        d[2] = (bf16_t)(f.z * scale); d[3] = (bf16_t)(f.w * scale);
    }

    // ---- double-buffer staging state (registers) ----
    float4 kreg[4], vreg[4];

    // ---- accumulators (C-frag layout: elem j -> row, lane -> col) ----
    v8f o[8];
#pragma unroll
    for (int n = 0; n < 8; ++n)
#pragma unroll
        for (int j = 0; j < 8; ++j) o[n][j] = 0.0f;
    v8f m, l;
#pragma unroll
    for (int j = 0; j < 8; ++j) { m[j] = -1.0e30f; l[j] = 0.0f; }

    const int mrow0 = w * 16;
    const int halfk = (lane & 16) ? 8 : 0;     // A-frag K offset for upper half-wave
    const int mlane = lane & 15;

    // ---- preload tile 0 into LDS buffer 0 ----
#pragma unroll
    for (int i = 0; i < 4; ++i) {
        const int lin = (t + i * 256) * 4;
        const int r = lin / D_, c = lin % D_;
        kreg[i] = *(const float4*)(kbase + (size_t)r * D_ + c);
        vreg[i] = *(const float4*)(vbase + (size_t)r * D_ + c);
    }
#pragma unroll
    for (int i = 0; i < 4; ++i) {
        const int lin = (t + i * 256) * 4;
        const int r = lin / D_, c = lin % D_;
        Kts[0][(c + 0) * KSTR + r] = (bf16_t)kreg[i].x;
        Kts[0][(c + 1) * KSTR + r] = (bf16_t)kreg[i].y;
        Kts[0][(c + 2) * KSTR + r] = (bf16_t)kreg[i].z;
        Kts[0][(c + 3) * KSTR + r] = (bf16_t)kreg[i].w;
        bf16_t* dv = &Vs[0][r * VSTR + c];
        dv[0] = (bf16_t)vreg[i].x; dv[1] = (bf16_t)vreg[i].y;
        dv[2] = (bf16_t)vreg[i].z; dv[3] = (bf16_t)vreg[i].w;
    }

    for (int kb = 0; kb < NT; ++kb) {
        const int cur = kb & 1;
        __syncthreads();   // buffer `cur` fully staged; prev reads of `1-cur` done

        // ---- issue next tile's global loads now (latency hides under WMMAs) ----
        if (kb + 1 < NT) {
#pragma unroll
            for (int i = 0; i < 4; ++i) {
                const int lin = (t + i * 256) * 4;
                const int r = lin / D_, c = lin % D_;
                kreg[i] = *(const float4*)(kbase + (size_t)((kb + 1) * BC + r) * D_ + c);
                vreg[i] = *(const float4*)(vbase + (size_t)((kb + 1) * BC + r) * D_ + c);
            }
        }
        if (kb + 2 < NT) {   // prefetch tile kb+2 toward L2 -> global_prefetch_b8
            __builtin_prefetch(kbase + ((size_t)(kb + 2) * BC) * D_ + t * 16, 0, 0);
            __builtin_prefetch(vbase + ((size_t)(kb + 2) * BC) * D_ + t * 16, 0, 0);
        }

        // ---- S = (Q*scale) @ K^T : 16x32 scores per wave, 8 WMMAs ----
        v8f sf[2];
#pragma unroll
        for (int n = 0; n < 2; ++n)
#pragma unroll
            for (int j = 0; j < 8; ++j) sf[n][j] = 0.0f;

#pragma unroll
        for (int kc = 0; kc < D_ / 32; ++kc) {
            const bf16_t* ap = &Qs[(mrow0 + mlane) * QSTR + kc * 32 + halfk];
            const v16bf a = pack16(*(const v8bf*)ap, *(const v8bf*)(ap + 16));
#pragma unroll
            for (int n = 0; n < 2; ++n) {
                const bf16_t* bp = &Kts[cur][(kc * 32 + halfk) * KSTR + n * 16 + mlane];
                v16bf bb;
#pragma unroll
                for (int i = 0; i < 8; ++i) { bb[i] = bp[i * KSTR]; bb[i + 8] = bp[(i + 16) * KSTR]; }
                sf[n] = __builtin_amdgcn_wmma_f32_16x16x32_bf16(
                    false, a, false, bb, (short)0, sf[n], false, false);
            }
        }

        // ---- online softmax (rows = frag elems; reduce over 16-lane half) ----
        v8f mt;
#pragma unroll
        for (int j = 0; j < 8; ++j) mt[j] = fmaxf(sf[0][j], sf[1][j]);
#pragma unroll
        for (int off = 8; off > 0; off >>= 1)
#pragma unroll
            for (int j = 0; j < 8; ++j)
                mt[j] = fmaxf(mt[j], __shfl_xor(mt[j], off, 16));

        v8f alpha, p0, p1, rs;
#pragma unroll
        for (int j = 0; j < 8; ++j) {
            const float mn = fmaxf(m[j], mt[j]);
            alpha[j] = __expf(m[j] - mn);
            p0[j] = __expf(sf[0][j] - mn);
            p1[j] = __expf(sf[1][j] - mn);
            m[j] = mn;
            rs[j] = p0[j] + p1[j];
        }
#pragma unroll
        for (int off = 8; off > 0; off >>= 1)
#pragma unroll
            for (int j = 0; j < 8; ++j) rs[j] += __shfl_xor(rs[j], off, 16);
#pragma unroll
        for (int j = 0; j < 8; ++j) l[j] = l[j] * alpha[j] + rs[j];
#pragma unroll
        for (int n = 0; n < 8; ++n)
#pragma unroll
            for (int j = 0; j < 8; ++j) o[n][j] *= alpha[j];

        // ---- P -> LDS (bf16, per-wave tile), re-load as A-fragment ----
        {
            const int rowoff = (lane & 16) ? 8 : 0;
            bf16_t* pb = &Ps[w][0];
#pragma unroll
            for (int j = 0; j < 8; ++j) {
                pb[(j + rowoff) * PSTR + mlane]      = (bf16_t)p0[j];
                pb[(j + rowoff) * PSTR + 16 + mlane] = (bf16_t)p1[j];
            }
        }
        const bf16_t* pap = &Ps[w][mlane * PSTR + halfk];
        const v16bf pa = pack16(*(const v8bf*)pap, *(const v8bf*)(pap + 16));

        // ---- O += P @ V : 8 WMMAs ----
#pragma unroll
        for (int n = 0; n < 8; ++n) {
            const bf16_t* vp = &Vs[cur][halfk * VSTR + n * 16 + mlane];
            v16bf bb;
#pragma unroll
            for (int i = 0; i < 8; ++i) { bb[i] = vp[i * VSTR]; bb[i + 8] = vp[(i + 16) * VSTR]; }
            o[n] = __builtin_amdgcn_wmma_f32_16x16x32_bf16(
                false, pa, false, bb, (short)0, o[n], false, false);
        }

        // ---- cvt + store next tile into the other LDS buffer ----
        if (kb + 1 < NT) {
            const int nxt = 1 - cur;
#pragma unroll
            for (int i = 0; i < 4; ++i) {
                const int lin = (t + i * 256) * 4;
                const int r = lin / D_, c = lin % D_;
                Kts[nxt][(c + 0) * KSTR + r] = (bf16_t)kreg[i].x;
                Kts[nxt][(c + 1) * KSTR + r] = (bf16_t)kreg[i].y;
                Kts[nxt][(c + 2) * KSTR + r] = (bf16_t)kreg[i].z;
                Kts[nxt][(c + 3) * KSTR + r] = (bf16_t)kreg[i].w;
                bf16_t* dv = &Vs[nxt][r * VSTR + c];
                dv[0] = (bf16_t)vreg[i].x; dv[1] = (bf16_t)vreg[i].y;
                dv[2] = (bf16_t)vreg[i].z; dv[3] = (bf16_t)vreg[i].w;
            }
        }
    }

    // ---- finalize: O /= l, sum over heads via f32 atomics ----
    v8f linv;
#pragma unroll
    for (int j = 0; j < 8; ++j) linv[j] = 1.0f / l[j];

    const int row0 = qblk * BR + mrow0 + ((lane & 16) ? 8 : 0);
    const int col0 = lane & 15;
#pragma unroll
    for (int n = 0; n < 8; ++n)
#pragma unroll
        for (int j = 0; j < 8; ++j)
            atomicAdd(&out[((size_t)b * S_ + row0 + j) * D_ + n * 16 + col0],
                      o[n][j] * linv[j]);
}

// ---------------------------------------------------------------------------
extern "C" void kernel_launch(void* const* d_in, const int* in_sizes, int n_in,
                              void* d_out, int out_size, void* d_ws, size_t ws_size,
                              hipStream_t stream) {
    const float* q = (const float*)d_in[0];
    const float* k = (const float*)d_in[1];
    const float* v = (const float*)d_in[2];
    float* out = (float*)d_out;
    float* ws  = (float*)d_ws;   // B*D floats

    zero_ws_kernel<<<1, B_ * D_, 0, stream>>>(ws);
    sumv_kernel<<<dim3(S_ / 256, H_, B_), 256, 0, stream>>>(v, ws);
    initout_kernel<<<(B_ * S_ * D_) / 256, 256, 0, stream>>>(ws, out);
    fa_bf16_wmma_kernel<<<dim3(S_ / BR, H_, B_), 256, 0, stream>>>(q, k, v, out);
}
